// DVS128GestureNet_62311385530930
// MI455X (gfx1250) — compile-verified, matrix operand order checked by reading, not run on hardware
//
#include <hip/hip_runtime.h>
#include <hip/hip_bf16.h>

typedef __attribute__((ext_vector_type(16))) _Float16 v16h;
typedef __attribute__((ext_vector_type(8)))  float    v8f;

#define BN_EPS 1e-5f

// ---- WMMA 16-bit fragment K mapping (CDNA5 ISA §7.12.2) ----
// lanes 0-15: VGPR0..3 = K{0..7}, VGPR4..7 = K{16..23}; lanes 16-31: +8.
// => per lane, fragment = 16B at (base + lh*16) and 16B at (base + 32 + lh*16)
__device__ __forceinline__ int frag_base(int q, int lh) {
  return ((q < 4) ? (2 * q) : (16 + 2 * (q - 4))) + (lh ? 8 : 0);
}
__device__ __forceinline__ int frag_k(int h, int lh) {
  return frag_base(h >> 1, lh) + (h & 1);
}

// ---- CDNA5 async global->LDS copy (ASYNCcnt tracked, CDNA5 ISA) --
__device__ __forceinline__ void async_load_b128(void* lds_ptr, const void* gptr) {
  unsigned lds_off = (unsigned)(size_t)lds_ptr; // low 32 bits of generic ptr = LDS offset
  unsigned long long ga = (unsigned long long)(size_t)gptr;
  asm volatile("global_load_async_to_lds_b128 %0, %1, off"
               :: "v"(lds_off), "v"(ga) : "memory");
}
__device__ __forceinline__ void wait_async0() {
  asm volatile("s_wait_asynccnt 0x0" ::: "memory");
}

// ---------------------------------------------------------------------------
__global__ __launch_bounds__(256) void zero_f32(float* p, unsigned long long n) {
  unsigned long long i = (unsigned long long)blockIdx.x * blockDim.x + threadIdx.x;
  unsigned long long stride = (unsigned long long)gridDim.x * blockDim.x;
  for (; i < n; i += stride) p[i] = 0.0f;
}

// ---------------------------------------------------------------------------
// weight prep: fold BN scale, convert to f16, swizzle to WMMA fragment order
// ---------------------------------------------------------------------------
__global__ __launch_bounds__(256) void prep_bias(const float* g, const float* b,
                                                 const float* m, const float* v,
                                                 float* bias) {
  int i = blockIdx.x * 256 + threadIdx.x;
  if (i < 5 * 64) {
    float inv = rsqrtf(v[i] + BN_EPS);
    bias[i] = b[i] - m[i] * g[i] * inv;
  }
}

// conv0: 64oc x K(=2ic*9tap, padded to 32). layout [mt4][lane32][h16]
__global__ __launch_bounds__(256) void prep_w0(const float* w0, const float* g,
                                               const float* var, _Float16* out) {
  int i = blockIdx.x * 256 + threadIdx.x;
  if (i >= 4 * 32 * 16) return;
  int h = i & 15, lane = (i >> 4) & 31, mt = i >> 9;
  int oc = mt * 16 + (lane & 15);
  int k = frag_k(h, lane >> 4);
  float val = 0.0f;
  if (k < 18) {
    int tap = k >> 1, ic = k & 1;
    int ky = tap / 3, kx = tap % 3;
    float scale = g[oc] * rsqrtf(var[oc] + BN_EPS);
    val = w0[((oc * 2 + ic) * 3 + ky) * 3 + kx] * scale;
  }
  out[i] = (_Float16)val;
}

// layers 1..4: layout per layer [tap9][kc2][mt4][lane32][h16]
__global__ __launch_bounds__(256) void prep_wl(const float* wl, const float* g,
                                               const float* var, _Float16* out) {
  int i = blockIdx.x * 256 + threadIdx.x;
  const int per = 9 * 2 * 4 * 32 * 16; // 36864
  if (i >= 4 * per) return;
  int layer = i / per, j = i % per;
  int h = j & 15, lane = (j >> 4) & 31, mt = (j >> 9) & 3, kc = (j >> 11) & 1, tap = j >> 12;
  int oc = mt * 16 + (lane & 15);
  int ic = kc * 32 + frag_k(h, lane >> 4);
  int ky = tap / 3, kx = tap % 3;
  int labs = layer + 1;
  float scale = g[labs * 64 + oc] * rsqrtf(var[labs * 64 + oc] + BN_EPS);
  float val = wl[(((layer * 64 + oc) * 64 + ic) * 3 + ky) * 3 + kx] * scale;
  out[i] = (_Float16)val;
}

// fc1: B fragments [kc32][nt16][lane32][h16]; permute cols NCHW-flat -> NHWC-flat
__global__ __launch_bounds__(256) void prep_w1(const float* w1, _Float16* out) {
  int i = blockIdx.x * 256 + threadIdx.x;
  if (i >= 32 * 16 * 32 * 16) return;
  int h = i & 15, lane = (i >> 4) & 31, nt = (i >> 9) & 15, kc = i >> 13;
  int oc = nt * 16 + (lane & 15);
  int k = kc * 32 + frag_k(h, lane >> 4); // our NHWC flatten: ((y*4+x)*64 + c)
  int c = k & 63, pix = k >> 6;
  int ref = c * 16 + pix;                 // reference flatten: c*16 + y*4 + x
  out[i] = (_Float16)w1[oc * 1024 + ref];
}

// fc2: [kc8][nt7][lane32][h16], oc padded 110 -> 112
__global__ __launch_bounds__(256) void prep_w2(const float* w2, _Float16* out) {
  int i = blockIdx.x * 256 + threadIdx.x;
  if (i >= 8 * 7 * 32 * 16) return;
  int kc = i / 3584, r = i % 3584;
  int nt = r >> 9, lane = (r >> 4) & 31, h = r & 15;
  int oc = nt * 16 + (lane & 15);
  int k = kc * 32 + frag_k(h, lane >> 4);
  float val = (oc < 110) ? w2[oc * 256 + k] : 0.0f;
  out[i] = (_Float16)val;
}

// ---------------------------------------------------------------------------
// conv layer 0: 2 -> 64 ch @ 128x128, K=18 (one padded WMMA K-chunk),
// fused BN bias + LIF + 2x2 maxpool. Block = 128 thr = 4 waves, tile 8x8 px.
// Membrane state in coalesced fragment-major layout [bid][wv][nt][r][lane].
// ---------------------------------------------------------------------------
__global__ __launch_bounds__(128) void conv0_kernel(
    const float* __restrict__ x, int t,
    const _Float16* __restrict__ w0s, const float* __restrict__ bias,
    float* __restrict__ v0, _Float16* __restrict__ s0) {
  __shared__ __align__(16) _Float16 patch[10][10][2];
  __shared__ __align__(16) _Float16 stile[8][8][64];
  const int tid = threadIdx.x, bid = blockIdx.x;
  const int tx = bid & 15, ty = (bid >> 4) & 15, n = bid >> 8;

  for (int i = tid; i < 200; i += 128) {
    int ic = i & 1, pi = i >> 1, px = pi % 10, py = pi / 10;
    int gy = ty * 8 + py - 1, gx = tx * 8 + px - 1;
    float v = 0.0f;
    if (gy >= 0 && gy < 128 && gx >= 0 && gx < 128)
      v = x[(((size_t)(n * 4 + t) * 2 + ic) << 14) + gy * 128 + gx];
    patch[py][px][ic] = (_Float16)v;
  }
  __syncthreads();

  const int wv = tid >> 5, lane = tid & 31, lh = lane >> 4, nl = lane & 15;
  const v16h a = *(const v16h*)&w0s[(wv * 32 + lane) << 4];
  v8f acc[4] = {};

  for (int nt = 0; nt < 4; ++nt) {
    int p = nt * 16 + nl, py = p >> 3, px = p & 7;
    union { v16h v; unsigned u[8]; } b;
    for (int q = 0; q < 8; ++q) {
      int base = frag_base(q, lh);
      int tap = base >> 1; // pair = (ic0, ic1) of one tap
      unsigned val = 0u;
      if (tap < 9) val = *(const unsigned*)&patch[py + tap / 3][px + tap % 3][0];
      b.u[q] = val;
    }
    acc[nt] = __builtin_amdgcn_wmma_f32_16x16x32_f16(false, a, false, b.v,
                                                     (short)0, acc[nt], false, false);
  }

  float* vp = v0 + ((size_t)bid * 4 + wv) * 1024 + lane; // [bid][wv][nt][r][lane]
  for (int nt = 0; nt < 4; ++nt) {
    int p = nt * 16 + nl, py = p >> 3, px = p & 7;
    for (int r = 0; r < 8; ++r) {
      int oc = wv * 16 + lh * 8 + r;
      float xv = acc[nt][r] + bias[oc];
      float* vpe = vp + (nt * 8 + r) * 32;
      float vm = 0.5f * (*vpe + xv);
      float s = (vm >= 1.0f) ? 1.0f : 0.0f;
      *vpe = (s != 0.0f) ? 0.0f : vm;
      stile[py][px][oc] = (_Float16)s;
    }
  }
  __syncthreads();

  for (int i = tid; i < 1024; i += 128) {
    int c = i & 63, pp = i >> 6, ppy = pp >> 2, ppx = pp & 3;
    float m0 = (float)stile[2 * ppy][2 * ppx][c];
    m0 = fmaxf(m0, (float)stile[2 * ppy][2 * ppx + 1][c]);
    m0 = fmaxf(m0, (float)stile[2 * ppy + 1][2 * ppx][c]);
    m0 = fmaxf(m0, (float)stile[2 * ppy + 1][2 * ppx + 1][c]);
    s0[((((size_t)n * 64 + (ty * 4 + ppy)) * 64) + (tx * 4 + ppx)) * 64 + c] = (_Float16)m0;
  }
}

// ---------------------------------------------------------------------------
// conv layers 1..4: 64 -> 64 ch, implicit GEMM over 9 taps x 2 K-chunks.
// Halo patch staged via CDNA5 async global->LDS b128; B fragments are
// 2x ds_load_b128; membrane state fragment-major coalesced.
// ---------------------------------------------------------------------------
__global__ __launch_bounds__(128) void convN_kernel(
    const _Float16* __restrict__ sp_in, const _Float16* __restrict__ wl,
    const float* __restrict__ bias, float* __restrict__ vbuf,
    _Float16* __restrict__ sp_out, int H, int W) {
  __shared__ __align__(16) _Float16 patch[10][10][64];
  __shared__ __align__(16) _Float16 stile[8][8][64];
  const int tid = threadIdx.x, bid = blockIdx.x;
  const int tilesX = W >> 3, tpi = tilesX * (H >> 3);
  const int n = bid / tpi, rem = bid % tpi;
  const int ty = rem / tilesX, tx = rem % tilesX;

  // stage 10x10x64 halo patch: async copies for in-range rows, zero-fill pad
  for (int i = tid; i < 800; i += 128) {
    int c8 = i & 7, pi = i >> 3, px = pi % 10, py = pi / 10;
    int gy = ty * 8 + py - 1, gx = tx * 8 + px - 1;
    void* dst = &patch[py][px][c8 * 8];
    if (gy >= 0 && gy < H && gx >= 0 && gx < W) {
      async_load_b128(dst, &sp_in[((((size_t)n * H + gy) * W + gx) << 6) + c8 * 8]);
    } else {
      uint4 z = {0u, 0u, 0u, 0u};
      *(uint4*)dst = z;
    }
  }
  wait_async0();
  __syncthreads();

  const int wv = tid >> 5, lane = tid & 31, lh = lane >> 4, nl = lane & 15;
  v8f acc[4] = {};
  int pys[4], pxs[4];
  for (int nt = 0; nt < 4; ++nt) { int p = nt * 16 + nl; pys[nt] = p >> 3; pxs[nt] = p & 7; }

  for (int tap = 0; tap < 9; ++tap) {
    const int ky = tap / 3, kx = tap % 3;
    for (int kc = 0; kc < 2; ++kc) {
      const v16h a = *(const v16h*)&wl[(((tap * 2 + kc) * 4 + wv) * 32 + lane) << 4];
      for (int nt = 0; nt < 4; ++nt) {
        union { v16h v; uint4 u4[2]; } b;
        const char* s = (const char*)&patch[pys[nt] + ky][pxs[nt] + kx][kc * 32] + lh * 16;
        b.u4[0] = *(const uint4*)(s);
        b.u4[1] = *(const uint4*)(s + 32);
        acc[nt] = __builtin_amdgcn_wmma_f32_16x16x32_f16(false, a, false, b.v,
                                                         (short)0, acc[nt], false, false);
      }
    }
  }

  float* vp = vbuf + ((size_t)bid * 4 + wv) * 1024 + lane; // [bid][wv][nt][r][lane]
  for (int nt = 0; nt < 4; ++nt) {
    for (int r = 0; r < 8; ++r) {
      int oc = wv * 16 + lh * 8 + r;
      float xv = acc[nt][r] + bias[oc];
      float* vpe = vp + (nt * 8 + r) * 32;
      float vm = 0.5f * (*vpe + xv);
      float s = (vm >= 1.0f) ? 1.0f : 0.0f;
      *vpe = (s != 0.0f) ? 0.0f : vm;
      stile[pys[nt]][pxs[nt]][oc] = (_Float16)s;
    }
  }
  __syncthreads();

  const int Hp = H >> 1, Wp = W >> 1;
  for (int i = tid; i < 1024; i += 128) {
    int c = i & 63, pp = i >> 6, ppy = pp >> 2, ppx = pp & 3;
    float m0 = (float)stile[2 * ppy][2 * ppx][c];
    m0 = fmaxf(m0, (float)stile[2 * ppy][2 * ppx + 1][c]);
    m0 = fmaxf(m0, (float)stile[2 * ppy + 1][2 * ppx][c]);
    m0 = fmaxf(m0, (float)stile[2 * ppy + 1][2 * ppx + 1][c]);
    sp_out[((((size_t)n * Hp + (ty * 4 + ppy)) * Wp) + (tx * 4 + ppx)) * 64 + c] = (_Float16)m0;
  }
}

// ---------------------------------------------------------------------------
// fc1: [8,1024] x [1024,256] spikes GEMM, batch padded to 16 rows, fused LIF
// ---------------------------------------------------------------------------
__global__ __launch_bounds__(128) void fc1_kernel(
    const _Float16* __restrict__ s4, const _Float16* __restrict__ w1s,
    float* __restrict__ v_fc1, _Float16* __restrict__ sp1) {
  const int tid = threadIdx.x, wv = tid >> 5, lane = tid & 31, lh = lane >> 4, nl = lane & 15;
  v8f acc[4] = {};
  for (int kc = 0; kc < 32; ++kc) {
    union { v16h v; uint4 u4[2]; } a;
    uint4 z = {0u, 0u, 0u, 0u};
    if (nl < 8) {
      const char* s = (const char*)(s4 + nl * 1024 + kc * 32) + lh * 16;
      a.u4[0] = *(const uint4*)(s);
      a.u4[1] = *(const uint4*)(s + 32);
    } else { a.u4[0] = z; a.u4[1] = z; }
    for (int j = 0; j < 4; ++j) {
      int nt = wv + j * 4;
      v16h b = *(const v16h*)&w1s[((kc * 16 + nt) * 32 + lane) << 4];
      acc[j] = __builtin_amdgcn_wmma_f32_16x16x32_f16(false, a.v, false, b,
                                                      (short)0, acc[j], false, false);
    }
  }
  if (lh == 0) { // rows 8..15 of padded batch are garbage
    for (int j = 0; j < 4; ++j) {
      int nt = wv + j * 4;
      int oc = nt * 16 + nl;
      for (int r = 0; r < 8; ++r) {
        float vm = 0.5f * (v_fc1[r * 256 + oc] + acc[j][r]);
        float s = (vm >= 1.0f) ? 1.0f : 0.0f;
        v_fc1[r * 256 + oc] = (s != 0.0f) ? 0.0f : vm;
        sp1[r * 256 + oc] = (_Float16)s;
      }
    }
  }
}

// ---------------------------------------------------------------------------
// fc2: [8,256] x [256,112] GEMM, fused LIF + boost avg-pool + accumulate out
// ---------------------------------------------------------------------------
__global__ __launch_bounds__(128) void fc2_kernel(
    const _Float16* __restrict__ sp1, const _Float16* __restrict__ w2s,
    float* __restrict__ v_fc2, float* __restrict__ out) {
  __shared__ float sout[8][112];
  const int tid = threadIdx.x, wv = tid >> 5, lane = tid & 31, lh = lane >> 4, nl = lane & 15;
  for (int i = tid; i < 8 * 112; i += 128) ((float*)sout)[i] = 0.0f;
  __syncthreads();

  v8f acc[2] = {};
  const int ntc = (wv < 3) ? 2 : 1; // 7 N-tiles over 4 waves
  for (int kc = 0; kc < 8; ++kc) {
    union { v16h v; uint4 u4[2]; } a;
    uint4 z = {0u, 0u, 0u, 0u};
    if (nl < 8) {
      const char* s = (const char*)(sp1 + nl * 256 + kc * 32) + lh * 16;
      a.u4[0] = *(const uint4*)(s);
      a.u4[1] = *(const uint4*)(s + 32);
    } else { a.u4[0] = z; a.u4[1] = z; }
    for (int j = 0; j < ntc; ++j) {
      int nt = wv + j * 4;
      v16h b = *(const v16h*)&w2s[((kc * 7 + nt) * 32 + lane) << 4];
      acc[j] = __builtin_amdgcn_wmma_f32_16x16x32_f16(false, a.v, false, b,
                                                      (short)0, acc[j], false, false);
    }
  }
  if (lh == 0) {
    for (int j = 0; j < ntc; ++j) {
      int nt = wv + j * 4;
      int oc = nt * 16 + nl;
      if (oc < 110) {
        for (int r = 0; r < 8; ++r) {
          float vm = 0.5f * (v_fc2[r * 112 + oc] + acc[j][r]);
          float s = (vm >= 1.0f) ? 1.0f : 0.0f;
          v_fc2[r * 112 + oc] = (s != 0.0f) ? 0.0f : vm;
          sout[r][oc] = s;
        }
      }
    }
  }
  __syncthreads();
  if (tid < 88) {
    int n = tid / 11, jj = tid % 11;
    float sum = 0.0f;
    for (int i = 0; i < 10; ++i) sum += sout[n][jj * 10 + i];
    out[n * 11 + jj] += 0.1f * sum;
  }
}

// ---------------------------------------------------------------------------
extern "C" void kernel_launch(void* const* d_in, const int* in_sizes, int n_in,
                              void* d_out, int out_size, void* d_ws, size_t ws_size,
                              hipStream_t stream) {
  const float* x        = (const float*)d_in[0];
  const float* conv0_w  = (const float*)d_in[1];
  const float* convs_w  = (const float*)d_in[2];
  const float* bn_gamma = (const float*)d_in[3];
  const float* bn_beta  = (const float*)d_in[4];
  const float* bn_mean  = (const float*)d_in[5];
  const float* bn_var   = (const float*)d_in[6];
  const float* fc1_w    = (const float*)d_in[7];
  const float* fc2_w    = (const float*)d_in[8];
  float* out = (float*)d_out;
  char* ws = (char*)d_ws;

  size_t off = 0;
  auto alloc = [&](size_t bytes) { size_t o = off; off = (off + bytes + 255) & ~(size_t)255; return o; };
  const size_t o_bias = alloc(5 * 64 * 4);
  const size_t o_w0s  = alloc(4 * 32 * 16 * 2);
  const size_t o_wls  = alloc(4 * 36864 * 2);
  const size_t o_w1s  = alloc(32 * 16 * 32 * 16 * 2);
  const size_t o_w2s  = alloc(8 * 7 * 32 * 16 * 2);
  const size_t o_s0   = alloc((size_t)8 * 64 * 64 * 64 * 2);
  const size_t o_s1   = alloc((size_t)8 * 32 * 32 * 64 * 2);
  const size_t o_s2   = alloc((size_t)8 * 16 * 16 * 64 * 2);
  const size_t o_s3   = alloc((size_t)8 * 8 * 8 * 64 * 2);
  const size_t o_s4   = alloc((size_t)8 * 4 * 4 * 64 * 2);
  const size_t o_sp1  = alloc(8 * 256 * 2);
  const size_t o_v0   = alloc((size_t)8 * 128 * 128 * 64 * 4);
  const size_t o_v1   = alloc((size_t)8 * 64 * 64 * 64 * 4);
  const size_t o_v2   = alloc((size_t)8 * 32 * 32 * 64 * 4);
  const size_t o_v3   = alloc((size_t)8 * 16 * 16 * 64 * 4);
  const size_t o_v4   = alloc((size_t)8 * 8 * 8 * 64 * 4);
  const size_t o_vfc1 = alloc(8 * 256 * 4);
  const size_t o_vfc2 = alloc(8 * 112 * 4);
  const size_t v_end  = off;
  if (v_end > ws_size) return; // workspace too small: bail deterministically

  float*    biasb = (float*)(ws + o_bias);
  _Float16* w0s   = (_Float16*)(ws + o_w0s);
  _Float16* wls   = (_Float16*)(ws + o_wls);
  _Float16* w1s   = (_Float16*)(ws + o_w1s);
  _Float16* w2s   = (_Float16*)(ws + o_w2s);
  _Float16* sbuf[5] = {(_Float16*)(ws + o_s0), (_Float16*)(ws + o_s1),
                       (_Float16*)(ws + o_s2), (_Float16*)(ws + o_s3),
                       (_Float16*)(ws + o_s4)};
  _Float16* sp1 = (_Float16*)(ws + o_sp1);
  float* vbuf[5] = {(float*)(ws + o_v0), (float*)(ws + o_v1), (float*)(ws + o_v2),
                    (float*)(ws + o_v3), (float*)(ws + o_v4)};
  float* vfc1 = (float*)(ws + o_vfc1);
  float* vfc2 = (float*)(ws + o_vfc2);

  // zero membrane state region (v0..v_fc2 contiguous) and the output
  zero_f32<<<2048, 256, 0, stream>>>((float*)(ws + o_v0), (v_end - o_v0) / 4);
  zero_f32<<<1, 256, 0, stream>>>(out, 88);

  // weight prep (BN fold + f16 swizzle into WMMA fragment order)
  prep_bias<<<2, 256, 0, stream>>>(bn_gamma, bn_beta, bn_mean, bn_var, biasb);
  prep_w0<<<8, 256, 0, stream>>>(conv0_w, bn_gamma, bn_var, w0s);
  prep_wl<<<576, 256, 0, stream>>>(convs_w, bn_gamma, bn_var, wls);
  prep_w1<<<1024, 256, 0, stream>>>(fc1_w, w1s);
  prep_w2<<<112, 256, 0, stream>>>(fc2_w, w2s);

  // sequential timesteps (LIF state dependency)
  for (int t = 0; t < 4; ++t) {
    conv0_kernel<<<2048, 128, 0, stream>>>(x, t, w0s, biasb, vbuf[0], sbuf[0]);
    for (int l = 1; l < 5; ++l) {
      int H = 128 >> l;
      int blocks = 8 * (H >> 3) * (H >> 3);
      convN_kernel<<<blocks, 128, 0, stream>>>(sbuf[l - 1], wls + (l - 1) * 36864,
                                               biasb + l * 64, vbuf[l], sbuf[l], H, H);
    }
    fc1_kernel<<<1, 128, 0, stream>>>(sbuf[4], w1s, vfc1, sp1);
    fc2_kernel<<<1, 128, 0, stream>>>(sp1, w2s, vfc2, out);
  }
}
